// TensorProductScoreModelAllAtom_28140625723919
// MI455X (gfx1250) — matrix operand order repaired and checked by reading, not compiled, not run on hardware
//
#include <hip/hip_runtime.h>
#include <hip/hip_bf16.h>

// ---------------------------------------------------------------------------
// CDNA5 / gfx1250 implementation of TensorProductScoreModelAllAtom layer.
// wave32, WMMA f32_16x16x32_bf16. One wave owns a 16-edge row strip.
// ---------------------------------------------------------------------------

typedef __attribute__((ext_vector_type(16))) __bf16 v16bf;
typedef __attribute__((ext_vector_type(8)))  __bf16 v8bf;
typedef __attribute__((ext_vector_type(8)))  float  v8f;

#define NS    48
#define HN    144      // 3*NS
#define ZK    160      // K padded to multiple of 32
#define WNP   2784     // NS*NS + NS*NV
#define NVv   10
#define OUTD  78       // NS + 3*NV
#define INVC  0.14433756729740643f   // 1/sqrt(48)

__device__ __forceinline__ unsigned short f2bf(float f) {
    unsigned int u = __float_as_uint(f);
    u += 0x7FFFu + ((u >> 16) & 1u);          // round-to-nearest-even
    return (unsigned short)(u >> 16);
}
__device__ __forceinline__ float bf2f(unsigned short u) {
    return __uint_as_float(((unsigned int)u) << 16);
}

__device__ __forceinline__ v16bf load_afrag(const unsigned short* arow, int kt, int hf) {
    v8bf lo = *(const v8bf*)(arow + kt * 32 + hf * 16);
    v8bf hi = *(const v8bf*)(arow + kt * 32 + hf * 16 + 8);
    return __builtin_shufflevector(lo, hi, 0,1,2,3,4,5,6,7,8,9,10,11,12,13,14,15);
}

// ---------------------------------------------------------------- utilities
__global__ void k_zero_u32(unsigned int* p, long long n) {
    long long i = (long long)blockIdx.x * blockDim.x + threadIdx.x;
    if (i < n) p[i] = 0u;
}

__global__ void k_out_init(const float* lig, const float* rec, const float* atom,
                           float* out, int nlig, int nrec, int natom) {
    long long tot = (long long)(nlig + nrec + natom) * OUTD;
    long long i = (long long)blockIdx.x * blockDim.x + threadIdx.x;
    if (i >= tot) return;
    int row = (int)(i / OUTD), col = (int)(i % OUTD);
    float v = 0.f;
    if (col < NS) {
        if (row < nlig)              v = lig[(long long)row * NS + col];
        else if (row < nlig + nrec)  v = rec[(long long)(row - nlig) * NS + col];
        else                         v = atom[(long long)(row - nlig - nrec) * NS + col];
    }
    out[i] = v;
}

// W1/W2 transpose+pad: wt[c][n][k] = w[c][k][n]  (bf16, k padded to ZK)
__global__ void k_prep_w12(const float* w, unsigned short* wt) {
    long long n = 9LL * HN * ZK;
    long long i = (long long)blockIdx.x * blockDim.x + threadIdx.x;
    if (i >= n) return;
    int c = (int)(i / (HN * ZK));
    int r = (int)(i % (HN * ZK));
    int nn = r / ZK, k = r % ZK;
    float v = (k < HN) ? w[((long long)c * HN + k) * HN + nn] : 0.f;
    wt[i] = f2bf(v);
}

// W3 permute+transpose+pad:
//   dest col n' = g*48 + i, g in [0,58):  g<48 -> src col = i*48+g (path 0e)
//                                          g>=48 -> src col = 2304 + i*10 + (g-48)
__global__ void k_prep_w3(const float* w3, const float* b3,
                          unsigned short* wt3, float* b3p) {
    long long n = 9LL * WNP * ZK;
    long long i = (long long)blockIdx.x * blockDim.x + threadIdx.x;
    if (i >= n) return;
    int c  = (int)(i / ((long long)WNP * ZK));
    int r  = (int)(i % ((long long)WNP * ZK));
    int np = r / ZK, k = r % ZK;
    int g = np / NS, ii = np % NS;
    int src = (g < NS) ? (ii * NS + g) : (NS * NS + ii * NVv + (g - NS));
    float v = (k < HN) ? w3[((long long)c * HN + k) * WNP + src] : 0.f;
    wt3[i] = f2bf(v);
    if (k == 0) b3p[(long long)c * WNP + np] = b3[(long long)c * WNP + src];
}

__global__ void k_deg(const int* dst, int E, float* deg) {
    int i = blockIdx.x * blockDim.x + threadIdx.x;
    if (i < E) atomicAdd(&deg[dst[i]], 1.f);
}

// Z[e] = bf16([attr(48), src_x[sidx](48), dst_x[didx](48), 0 pad]), rows [E,Epad)=0
__global__ void k_zbuild(const float* attr, const float* sx, const float* dx,
                         const int* sidx, const int* didx, int E, int Epad,
                         unsigned short* Z) {
    long long n = (long long)Epad * ZK;
    long long i = (long long)blockIdx.x * blockDim.x + threadIdx.x;
    if (i >= n) return;
    int e = (int)(i / ZK), k = (int)(i % ZK);
    float v = 0.f;
    if (e < E) {
        if (k < NS)            v = attr[(long long)e * NS + k];
        else if (k < 2 * NS)   v = sx[(long long)sidx[e] * NS + (k - NS)];
        else if (k < 3 * NS)   v = dx[(long long)didx[e] * NS + (k - 2 * NS)];
    }
    Z[i] = f2bf(v);
}

// ------------------------------------------------- GEMM + bias + ReLU (bf16)
// grid = (Mtiles). One wave per block computes the full 16x144 row strip:
// A fragments loaded once, 9 N-tiles x 5 K-steps = 45 WMMAs per wave.
// B fragments for a tile are loaded as a batch (clause) before the WMMA chain
// so loads of the next tile overlap the current tile's WMMAs.
__global__ __launch_bounds__(32)
void k_gemm_relu(const unsigned short* __restrict__ A,   // Epad x ZK bf16
                 const unsigned short* __restrict__ Bt,  // HN x ZK bf16 (n-major)
                 const float* __restrict__ bias,         // HN
                 unsigned short* __restrict__ H) {       // Epad x ZK bf16 (cols<144)
    int lane = threadIdx.x;
    int m  = lane & 15;
    int hf = lane >> 4;
    const unsigned short* arow = A + (long long)(blockIdx.x * 16 + m) * ZK;
    v16bf afr[5];
#pragma unroll
    for (int kt = 0; kt < 5; ++kt) afr[kt] = load_afrag(arow, kt, hf);

#pragma unroll 3
    for (int nt = 0; nt < HN / 16; ++nt) {
        const unsigned short* brow = Bt + (long long)(nt * 16 + m) * ZK + hf * 16;
        v16bf bfr[5];
#pragma unroll
        for (int kt = 0; kt < 5; ++kt) bfr[kt] = *(const v16bf*)(brow + kt * 32);
        v8f c;
#pragma unroll
        for (int j = 0; j < 8; ++j) c[j] = 0.f;
#pragma unroll
        for (int kt = 0; kt < 5; ++kt)
            c = __builtin_amdgcn_wmma_f32_16x16x32_bf16(false, afr[kt], false, bfr[kt],
                                                        (short)0, c, false, false);
        float bi = bias[nt * 16 + m];
#pragma unroll
        for (int r = 0; r < 8; ++r) {
            float v = c[r] + bi;
            v = v > 0.f ? v : 0.f;
            int row = blockIdx.x * 16 + r + hf * 8;
            H[(long long)row * ZK + nt * 16 + m] = f2bf(v);
        }
    }
}

// ------------------- fused layer3 GEMM + per-edge TP contraction + scatter
// One wave per 16-edge tile. For each output slot g (58 of them) the 48 i's
// span 3 consecutive 16-wide N-tiles of permuted W3: accumulate the three
// fs-weighted sub-tiles in registers, then one 16-lane butterfly per g.
__global__ __launch_bounds__(32)
void k_conv3(const unsigned short* __restrict__ H2,   // Epad x ZK bf16
             const unsigned short* __restrict__ Zb,   // Epad x ZK bf16 (fs @ 48..95)
             const unsigned short* __restrict__ W3T,  // WNP x ZK bf16
             const float* __restrict__ b3p,           // WNP
             const float* __restrict__ sh,            // E x 9
             const int* __restrict__ didx,
             const float* __restrict__ deg,
             float* __restrict__ out, int out_off, int E) {
    __shared__ float fs_s[16][NS];
    __shared__ float acc0[16][NS];
    __shared__ float accv[16][NVv];

    int lane = threadIdx.x;
    int mt = blockIdx.x;
    int m  = lane & 15;
    int hf = lane >> 4;

    for (int i = lane; i < 16 * NS; i += 32) {
        int r = i / NS, cc = i % NS;
        fs_s[r][cc] = bf2f(Zb[(long long)(mt * 16 + r) * ZK + NS + cc]);
        acc0[r][cc] = 0.f;
    }
    for (int i = lane; i < 16 * NVv; i += 32) accv[i / NVv][i % NVv] = 0.f;
    __syncthreads();

    // preload A fragments of this edge tile (16 x 160)
    const unsigned short* arow = H2 + (long long)(mt * 16 + m) * ZK;
    v16bf afr[5];
#pragma unroll
    for (int kt = 0; kt < 5; ++kt) afr[kt] = load_afrag(arow, kt, hf);

    // per-lane fs values: fsr[sub][r] = fs[row = r+8*hf][sub*16 + m]
    float fsr[3][8];
#pragma unroll
    for (int sub = 0; sub < 3; ++sub)
#pragma unroll
        for (int r = 0; r < 8; ++r)
            fsr[sub][r] = fs_s[r + hf * 8][sub * 16 + m];

    for (int g = 0; g < WNP / NS; ++g) {               // 58 output slots
        float vs[8];
#pragma unroll
        for (int r = 0; r < 8; ++r) vs[r] = 0.f;
#pragma unroll
        for (int sub = 0; sub < 3; ++sub) {
            int t = g * 3 + sub;
            const unsigned short* brow = W3T + (long long)(t * 16 + m) * ZK + hf * 16;
            v16bf bfr[5];
#pragma unroll
            for (int kt = 0; kt < 5; ++kt) bfr[kt] = *(const v16bf*)(brow + kt * 32);
            v8f c;
#pragma unroll
            for (int j = 0; j < 8; ++j) c[j] = 0.f;
#pragma unroll
            for (int kt = 0; kt < 5; ++kt)
                c = __builtin_amdgcn_wmma_f32_16x16x32_bf16(false, afr[kt], false, bfr[kt],
                                                            (short)0, c, false, false);
            float bi = b3p[t * 16 + m];
#pragma unroll
            for (int r = 0; r < 8; ++r)
                vs[r] += (c[r] + bi) * fsr[sub][r];
        }
#pragma unroll
        for (int r = 0; r < 8; ++r) {
            float v = vs[r];
            v += __shfl_xor(v, 1, 16);
            v += __shfl_xor(v, 2, 16);
            v += __shfl_xor(v, 4, 16);
            v += __shfl_xor(v, 8, 16);                 // sum over 48 i's (3 subs folded)
            if (m == 0) {
                int row = r + hf * 8;
                if (g < NS) acc0[row][g] += v;
                else        accv[row][g - NS] += v;
            }
        }
    }
    __syncthreads();

    // SH scaling, mean (1/deg), scatter-add into output
    for (int i = lane; i < 16 * OUTD; i += 32) {
        int r = i / OUTD, col = i % OUTD;
        int e = mt * 16 + r;
        if (e < E) {
            int d = didx[e];
            float dg = deg[d]; if (dg < 1.f) dg = 1.f;
            float s = INVC / dg;
            float v;
            if (col < NS) v = acc0[r][col] * sh[(long long)e * 9 + 0];
            else {
                int q = col - NS;
                v = accv[r][q / 3] * sh[(long long)e * 9 + 1 + q % 3];
            }
            atomicAdd(&out[(long long)(out_off + d) * OUTD + col], v * s);
        }
    }
}

// ---------------------------------------------------------------------------
extern "C" void kernel_launch(void* const* d_in, const int* in_sizes, int n_in,
                              void* d_out, int out_size, void* d_ws, size_t ws_size,
                              hipStream_t stream) {
    (void)n_in; (void)out_size; (void)ws_size;
    const float* lig  = (const float*)d_in[0];
    const float* rec  = (const float*)d_in[1];
    const float* atom = (const float*)d_in[2];
    const float* attrs[6] = { (const float*)d_in[3], (const float*)d_in[4],
                              (const float*)d_in[5], (const float*)d_in[6],
                              (const float*)d_in[7], (const float*)d_in[8] };
    const float* shs[6]   = { (const float*)d_in[9], (const float*)d_in[10],
                              (const float*)d_in[11], (const float*)d_in[12],
                              (const float*)d_in[13], (const float*)d_in[14] };
    const float* mw1 = (const float*)d_in[15];
    const float* mb1 = (const float*)d_in[16];
    const float* mw2 = (const float*)d_in[17];
    const float* mb2 = (const float*)d_in[18];
    const float* mw3 = (const float*)d_in[19];
    const float* mb3 = (const float*)d_in[20];
    const int* ll_src = (const int*)d_in[21]; const int* ll_dst = (const int*)d_in[22];
    const int* rr_src = (const int*)d_in[23]; const int* rr_dst = (const int*)d_in[24];
    const int* aa_src = (const int*)d_in[25]; const int* aa_dst = (const int*)d_in[26];
    const int* lr_lig = (const int*)d_in[27]; const int* lr_rec = (const int*)d_in[28];
    const int* la_lig = (const int*)d_in[29]; const int* la_atom= (const int*)d_in[30];
    const int* ar_atom= (const int*)d_in[31]; const int* ar_rec = (const int*)d_in[32];

    const int NLIG = in_sizes[0] / NS, NREC = in_sizes[1] / NS, NATOM = in_sizes[2] / NS;
    const int E_LL = in_sizes[3] / NS, E_RR = in_sizes[4] / NS, E_AA = in_sizes[5] / NS;
    const int E_LR = in_sizes[6] / NS, E_LA = in_sizes[7] / NS, E_AR = in_sizes[8] / NS;

    struct Conv {
        const float *attr, *sh, *sx, *dx; const int *sidx, *didx;
        int E, ndst, off;
    } cv[9] = {
        { attrs[0], shs[0], lig,  lig,  ll_src,  ll_dst,  E_LL, NLIG,  0 },
        { attrs[3], shs[3], rec,  lig,  lr_rec,  lr_lig,  E_LR, NLIG,  0 },
        { attrs[4], shs[4], atom, lig,  la_atom, la_lig,  E_LA, NLIG,  0 },
        { attrs[3], shs[3], lig,  rec,  lr_lig,  lr_rec,  E_LR, NREC,  NLIG },
        { attrs[1], shs[1], rec,  rec,  rr_src,  rr_dst,  E_RR, NREC,  NLIG },
        { attrs[5], shs[5], atom, rec,  ar_atom, ar_rec,  E_AR, NREC,  NLIG },
        { attrs[4], shs[4], lig,  atom, la_lig,  la_atom, E_LA, NATOM, NLIG + NREC },
        { attrs[5], shs[5], rec,  atom, ar_rec,  ar_atom, E_AR, NATOM, NLIG + NREC },
        { attrs[2], shs[2], atom, atom, aa_src,  aa_dst,  E_AA, NATOM, NLIG + NREC },
    };

    int epmax = 0;
    for (int c = 0; c < 9; ++c) {
        int ep = (cv[c].E + 15) & ~15;
        if (ep > epmax) epmax = ep;
    }

    // -------- workspace layout (bf16 region then f32 region, 32B aligned)
    unsigned short* wsu = (unsigned short*)d_ws;
    const long long W12 = 9LL * HN * ZK;          // 207360
    const long long W3  = 9LL * WNP * ZK;         // 4008960
    const long long ZH  = (long long)epmax * ZK;  // per buffer
    unsigned short* W1T = wsu;
    unsigned short* W2T = W1T + W12;
    unsigned short* W3T = W2T + W12;
    unsigned short* Zb  = W3T + W3;
    unsigned short* H1  = Zb + ZH;
    unsigned short* H2  = H1 + ZH;
    long long fbase = ((2 * W12 + W3 + 3 * ZH) * 2 + 255) & ~255LL;   // bytes
    float* b3p = (float*)((char*)d_ws + fbase);
    float* deg = b3p + 9LL * WNP;

    const int T = 256;
    auto blks = [](long long n, int t) { return (unsigned)((n + t - 1) / t); };

    // zero H1+H2 (pad cols 144..159 must stay 0 for all convs): 2*ZH u16 = ZH u32
    k_zero_u32<<<blks(ZH, T), T, 0, stream>>>((unsigned int*)H1, ZH);
    // residual init of output
    k_out_init<<<blks((long long)(NLIG + NREC + NATOM) * OUTD, T), T, 0, stream>>>(
        lig, rec, atom, (float*)d_out, NLIG, NREC, NATOM);
    // weight prep
    k_prep_w12<<<blks(W12, T), T, 0, stream>>>(mw1, W1T);
    k_prep_w12<<<blks(W12, T), T, 0, stream>>>(mw2, W2T);
    k_prep_w3 <<<blks(W3, T),  T, 0, stream>>>(mw3, mb3, W3T, b3p);

    for (int c = 0; c < 9; ++c) {
        const Conv& v = cv[c];
        int Epad = (v.E + 15) & ~15;
        int Mt = Epad / 16;
        k_zero_u32<<<blks(v.ndst, T), T, 0, stream>>>((unsigned int*)deg, v.ndst);
        k_deg<<<blks(v.E, T), T, 0, stream>>>(v.didx, v.E, deg);
        k_zbuild<<<blks((long long)Epad * ZK, T), T, 0, stream>>>(
            v.attr, v.sx, v.dx, v.sidx, v.didx, v.E, Epad, Zb);
        k_gemm_relu<<<Mt, 32, 0, stream>>>(
            Zb, W1T + (long long)c * HN * ZK, mb1 + c * HN, H1);
        k_gemm_relu<<<Mt, 32, 0, stream>>>(
            H1, W2T + (long long)c * HN * ZK, mb2 + c * HN, H2);
        k_conv3<<<Mt, 32, 0, stream>>>(
            H2, Zb, W3T + (long long)c * WNP * ZK, b3p + (long long)c * WNP,
            v.sh, v.didx, deg, (float*)d_out, v.off, v.E);
    }
}
